// VisionPart_25726854103530
// MI455X (gfx1250) — matrix operand order repaired and checked by reading, not compiled, not run on hardware
//
#include <hip/hip_runtime.h>
#include <cstdint>
#include <cstddef>

#define DEV __device__ __forceinline__

typedef __attribute__((ext_vector_type(16))) __bf16 v16bf;
typedef __attribute__((ext_vector_type(8)))  float  v8f;

static constexpr int Bv = 64, Nv = 576, Dv = 768, Lv = 64, Hv = 8, HDv = 96;
static constexpr float kScale    = 0.102062072615966f;   // 96^-0.5
static constexpr float kInvSqrtD = 0.0360843918243516f;  // 768^-0.5

#define WMMA_BF16(A, Bm, C) \
  __builtin_amdgcn_wmma_f32_16x16x32_bf16(false, (A), false, (Bm), (short)0, (C), false, false)

// CDNA5 async global->LDS copy (16B per lane), ASYNCcnt-tracked.
// vdst = LDS byte address (low 32 bits of generic shared pointer), vaddr = 64-bit global addr.
DEV void async_ld_b128(const void* gaddr, void* lds) {
  const unsigned la = (unsigned)(uintptr_t)lds;
  const unsigned long long ga = (unsigned long long)(uintptr_t)gaddr;
  asm volatile("global_load_async_to_lds_b128 %0, %1, off" :: "v"(la), "v"(ga) : "memory");
}
DEV void async_ld_b128_off16(const void* gaddr, void* lds) {
  const unsigned la = (unsigned)(uintptr_t)lds;
  const unsigned long long ga = (unsigned long long)(uintptr_t)gaddr;
  asm volatile("global_load_async_to_lds_b128 %0, %1, off offset:16" :: "v"(la), "v"(ga) : "memory");
}
DEV void wait_async0() { asm volatile("s_wait_asynccnt 0x0" ::: "memory"); }

DEV __bf16 f2bf(float f) {
  unsigned u = __float_as_uint(f);
  u += 0x7FFFu + ((u >> 16) & 1u);
  unsigned short h = (unsigned short)(u >> 16);
  return __builtin_bit_cast(__bf16, h);
}

DEV float geluf(float v) { return 0.5f * v * (1.0f + erff(v * 0.70710678118654752f)); }

DEV v8f zero8() { v8f z = {0.f,0.f,0.f,0.f,0.f,0.f,0.f,0.f}; return z; }

// A fragment 16x32 bf16: lanes 0-15 -> M=lane, K {0..7}U{16..23}; lanes 16-31 -> K {8..15}U{24..31}
DEV v16bf frag_a(const __bf16* s, int lda) {
  const int lane = threadIdx.x & 31;
  const int m = lane & 15;
  const int kh = (lane >> 4) << 3;
  v16bf a;
#pragma unroll
  for (int e = 0; e < 8; ++e) a[e] = s[m * lda + kh + e];
#pragma unroll
  for (int e = 0; e < 8; ++e) a[8 + e] = s[m * lda + 16 + kh + e];
  return a;
}

// B fragment 32x16 bf16 (KxN): lanes 0-15 -> N=lane, K {0..7}U{16..23}; lanes 16-31 -> K {8..15}U{24..31}
DEV v16bf frag_b(const __bf16* s, int ldb) {
  const int lane = threadIdx.x & 31;
  const int n = lane & 15;
  const int kh = (lane >> 4) << 3;
  v16bf b;
#pragma unroll
  for (int e = 0; e < 8; ++e) b[e] = s[(kh + e) * ldb + n];
#pragma unroll
  for (int e = 0; e < 8; ++e) b[8 + e] = s[(16 + kh + e) * ldb + n];
  return b;
}

DEV float block_sum(float v, float* red) {
  const int tid = threadIdx.x;
  red[tid] = v; __syncthreads();
  for (int s = 128; s > 0; s >>= 1) {
    if (tid < s) red[tid] += red[tid + s];
    __syncthreads();
  }
  float r = red[0]; __syncthreads();
  return r;
}

// ---------------- LayerNorm 1 + token score + sum(xn^2) ----------------
__launch_bounds__(256)
__global__ void ln1_kernel(const float* __restrict__ x, const float* __restrict__ g,
                           const float* __restrict__ bt, const float* __restrict__ sw,
                           const float* __restrict__ sb, float* __restrict__ xn,
                           __bf16* __restrict__ xnb, float* __restrict__ sq,
                           float* __restrict__ ts) {
  __shared__ float red[256];
  const int t = blockIdx.x, tid = threadIdx.x;
  const float* xp = x + (size_t)t * Dv;
  float v[3], s1 = 0.f, s2 = 0.f;
#pragma unroll
  for (int c = 0; c < 3; ++c) { v[c] = xp[tid + 256 * c]; s1 += v[c]; s2 += v[c] * v[c]; }
  s1 = block_sum(s1, red);
  s2 = block_sum(s2, red);
  const float mean = s1 * (1.0f / Dv);
  const float var  = s2 * (1.0f / Dv) - mean * mean;
  const float rstd = rsqrtf(var + 1e-5f);
  float accS = 0.f, accQ = 0.f;
#pragma unroll
  for (int c = 0; c < 3; ++c) {
    const int d = tid + 256 * c;
    const float y = (v[c] - mean) * rstd * g[d] + bt[d];
    xn[(size_t)t * Dv + d]  = y;
    xnb[(size_t)t * Dv + d] = f2bf(y);
    accS += y * sw[d];
    accQ += y * y;
  }
  accS = block_sum(accS, red);
  accQ = block_sum(accQ, red);
  if (tid == 0) { ts[t] = accS + sb[0]; sq[t] = accQ; }
}

// ---------------- LayerNorm 2 (bf16 out only) ----------------
__launch_bounds__(256)
__global__ void ln2_kernel(const float* __restrict__ x, const float* __restrict__ g,
                           const float* __restrict__ bt, __bf16* __restrict__ ob) {
  __shared__ float red[256];
  const int t = blockIdx.x, tid = threadIdx.x;
  const float* xp = x + (size_t)t * Dv;
  float v[3], s1 = 0.f, s2 = 0.f;
#pragma unroll
  for (int c = 0; c < 3; ++c) { v[c] = xp[tid + 256 * c]; s1 += v[c]; s2 += v[c] * v[c]; }
  s1 = block_sum(s1, red);
  s2 = block_sum(s2, red);
  const float mean = s1 * (1.0f / Dv);
  const float rstd = rsqrtf(s2 * (1.0f / Dv) - mean * mean + 1e-5f);
#pragma unroll
  for (int c = 0; c < 3; ++c) {
    const int d = tid + 256 * c;
    ob[(size_t)t * Dv + d] = f2bf((v[c] - mean) * rstd * g[d] + bt[d]);
  }
}

// ---------------- fp32 -> bf16 ----------------
__global__ void cvt_bf16_kernel(const float* __restrict__ s, __bf16* __restrict__ d, int n) {
  const int i = blockIdx.x * 256 + threadIdx.x;
  if (i < n) d[i] = f2bf(s[i]);
}

__global__ void init_kernel(unsigned* __restrict__ dmax) {
  if (threadIdx.x < Bv) dmax[threadIdx.x] = 0u;
}

// ---------------- pairwise distance: dm = sqrt(max(si+sj-2*fi.fj,0))/sqrt(D) ----------------
__launch_bounds__(256)
__global__ void pairdist_kernel(const __bf16* __restrict__ F, const float* __restrict__ sq,
                                float* __restrict__ dm) {
  __shared__ __align__(16) __bf16 As[64 * 32];
  __shared__ __align__(16) __bf16 Bs[32 * 64];
  const int tid = threadIdx.x, lane = tid & 31, w = tid >> 5;
  const int b = blockIdx.z, i0 = blockIdx.y * 64, j0 = blockIdx.x * 64;
  const __bf16* Fb = F + (size_t)b * Nv * Dv;
  const int ar = tid >> 2, ac = (tid & 3) * 8;
  const int mrow = (w >> 1) * 16, ncol = (w & 1) * 32;
  v8f acc0 = zero8(), acc1 = zero8();
  for (int k0 = 0; k0 < Dv; k0 += 32) {
    *(uint4*)&As[ar * 32 + ac] = *(const uint4*)&Fb[(size_t)(i0 + ar) * Dv + k0 + ac];
    uint4 t = *(const uint4*)&Fb[(size_t)(j0 + ar) * Dv + k0 + ac];
    const __bf16* te = reinterpret_cast<const __bf16*>(&t);
#pragma unroll
    for (int e = 0; e < 8; ++e) Bs[(ac + e) * 64 + ar] = te[e];
    __syncthreads();
    v16bf a  = frag_a(&As[mrow * 32], 32);
    v16bf b0 = frag_b(&Bs[ncol], 64);
    v16bf b1 = frag_b(&Bs[ncol + 16], 64);
    acc0 = WMMA_BF16(a, b0, acc0);
    acc1 = WMMA_BF16(a, b1, acc1);
    __syncthreads();
  }
  const int cn = lane & 15, cm = (lane >> 4) * 8;
#pragma unroll
  for (int r = 0; r < 8; ++r) {
    const int i = i0 + mrow + cm + r;
    const float si = sq[b * Nv + i];
    const int j0c = j0 + ncol + cn;
    const float d0 = si + sq[b * Nv + j0c]      - 2.f * acc0[r];
    const float d1 = si + sq[b * Nv + j0c + 16] - 2.f * acc1[r];
    float* row = dm + ((size_t)b * Nv + i) * Nv;
    row[j0c]      = sqrtf(fmaxf(d0, 0.f)) * kInvSqrtD;
    row[j0c + 16] = sqrtf(fmaxf(d1, 0.f)) * kInvSqrtD;
  }
}

// ---------------- kNN density (5 smallest dists incl. self) + per-batch dm max ----------------
__global__ void density_kernel(const float* __restrict__ dm, const float* __restrict__ noise,
                               float* __restrict__ dens, unsigned* __restrict__ dmax) {
  const int i = blockIdx.x * 256 + threadIdx.x;
  if (i >= Bv * Nv) return;
  const int b = i / Nv;
  const float* row = dm + (size_t)i * Nv;
  float b0 = 3e38f, b1 = 3e38f, b2 = 3e38f, b3 = 3e38f, b4 = 3e38f, rmax = 0.f;
  for (int j = 0; j < Nv; ++j) {
    const float d = row[j];
    rmax = fmaxf(rmax, d);
    if (d < b4) {
      if (d < b0)      { b4 = b3; b3 = b2; b2 = b1; b1 = b0; b0 = d; }
      else if (d < b1) { b4 = b3; b3 = b2; b2 = b1; b1 = d; }
      else if (d < b2) { b4 = b3; b3 = b2; b2 = d; }
      else if (d < b3) { b4 = b3; b3 = d; }
      else             { b4 = d; }
    }
  }
  const float msq = (b0*b0 + b1*b1 + b2*b2 + b3*b3 + b4*b4) * 0.2f;
  dens[i] = expf(-msq) + noise[i] * 1e-6f;
  atomicMax(dmax + b, __float_as_uint(rmax));  // order-invariant, deterministic
}

// ---------------- parent distance * density ----------------
__global__ void parent_kernel(const float* __restrict__ dm, const float* __restrict__ dens,
                              const unsigned* __restrict__ dmax, float* __restrict__ score) {
  const int i = blockIdx.x * 256 + threadIdx.x;
  if (i >= Bv * Nv) return;
  const int b = i / Nv;
  const float di = dens[i];
  const float* row = dm + (size_t)i * Nv;
  const float* db = dens + b * Nv;
  float best = __uint_as_float(dmax[b]);
  for (int j = 0; j < Nv; ++j)
    if (db[j] > di) best = fminf(best, row[j]);
  score[i] = best * di;
}

// ---------------- top-L by score (iterative argmax, tie -> smallest index) ----------------
__launch_bounds__(256)
__global__ void topl_kernel(const float* __restrict__ score, int* __restrict__ idx_down) {
  __shared__ float sc[Nv];
  __shared__ float rv[256];
  __shared__ int   ri[256];
  const int b = blockIdx.x, tid = threadIdx.x;
  for (int j = tid; j < Nv; j += 256) sc[j] = score[b * Nv + j];
  __syncthreads();
  for (int l = 0; l < Lv; ++l) {
    float bv = -3e38f; int bi = 0;
    for (int j = tid; j < Nv; j += 256)
      if (sc[j] > bv) { bv = sc[j]; bi = j; }
    rv[tid] = bv; ri[tid] = bi; __syncthreads();
    for (int s = 128; s > 0; s >>= 1) {
      if (tid < s) {
        if (rv[tid + s] > rv[tid] || (rv[tid + s] == rv[tid] && ri[tid + s] < ri[tid])) {
          rv[tid] = rv[tid + s]; ri[tid] = ri[tid + s];
        }
      }
      __syncthreads();
    }
    if (tid == 0) { idx_down[b * Lv + l] = ri[0]; sc[ri[0]] = -3e38f; }
    __syncthreads();
  }
}

// ---------------- assign each token to nearest center ----------------
__global__ void assign_kernel(const float* __restrict__ dm, const int* __restrict__ idx_down,
                              int* __restrict__ idxc) {
  const int i = blockIdx.x * 256 + threadIdx.x;
  if (i >= Bv * Nv) return;
  const int b = i / Nv, n = i % Nv;
  float bd = 3e38f; int bl = 0;
  for (int l = 0; l < Lv; ++l) {
    const int c = idx_down[b * Lv + l];
    const float d = dm[((size_t)b * Nv + c) * Nv + n];
    if (d < bd) { bd = d; bl = l; }
  }
  idxc[i] = bl;
}

__global__ void center_override_kernel(const int* __restrict__ idx_down, int* __restrict__ idxc) {
  const int i = blockIdx.x * 256 + threadIdx.x;
  if (i >= Bv * Lv) return;
  const int b = i / Lv, l = i % Lv;
  idxc[b * Nv + idx_down[i]] = l;
}

// ---------------- token merge (gather form; fixed reduction order) ----------------
__launch_bounds__(256)
__global__ void merge_kernel(const float* __restrict__ xn, const int* __restrict__ idxc,
                             const float* __restrict__ ts, float* __restrict__ qf,
                             __bf16* __restrict__ qb) {
  __shared__ float red[256];
  __shared__ float aws;
  const int l = blockIdx.x, b = blockIdx.y, tid = threadIdx.x;
  float p = 0.f;
  for (int n = tid; n < Nv; n += 256)
    if (idxc[b * Nv + n] == l) p += expf(ts[b * Nv + n]);
  p = block_sum(p, red);
  if (tid == 0) aws = p + 1e-6f;
  __syncthreads();
  const float aw = aws;
  float a0 = 0.f, a1 = 0.f, a2 = 0.f;
  for (int n = 0; n < Nv; ++n) {
    if (idxc[b * Nv + n] == l) {
      const float wgt = expf(ts[b * Nv + n]) / aw;
      const float* xp = xn + ((size_t)b * Nv + n) * Dv;
      a0 += xp[tid] * wgt; a1 += xp[tid + 256] * wgt; a2 += xp[tid + 512] * wgt;
    }
  }
  float* qp = qf + ((size_t)b * Lv + l) * Dv;
  qp[tid] = a0; qp[tid + 256] = a1; qp[tid + 512] = a2;
  __bf16* qpb = qb + ((size_t)b * Lv + l) * Dv;
  qpb[tid] = f2bf(a0); qpb[tid + 256] = f2bf(a1); qpb[tid + 512] = f2bf(a2);
}

// ---------------- generic bf16 WMMA GEMM: C[M,N] = A[M,K] x W[K,N] (+bias)(+gelu) ----------------
// BK=64 slab staged via CDNA5 async global->LDS; 4 WMMAs per barrier pair.
__launch_bounds__(256)
__global__ void gemm_bf16_kernel(const __bf16* __restrict__ A, const __bf16* __restrict__ W,
                                 const float* __restrict__ bias, float* __restrict__ Cf,
                                 __bf16* __restrict__ Cb, int M, int K, int N, int act) {
  __shared__ __align__(16) __bf16 As[64 * 64];
  __shared__ __align__(16) __bf16 Bs[64 * 64];
  const int tid = threadIdx.x, lane = tid & 31, w = tid >> 5;
  const int m0 = blockIdx.y * 64, n0 = blockIdx.x * 64;
  const int mrow = (w >> 1) * 16, ncol = (w & 1) * 32;
  const int ar = tid >> 2, ac = (tid & 3) * 16;   // 64x64 tile, 32B per thread
  v8f acc0 = zero8(), acc1 = zero8();
  for (int k0 = 0; k0 < K; k0 += 64) {
    const __bf16* ga = &A[(size_t)(m0 + ar) * K + k0 + ac];
    const __bf16* gw = &W[(size_t)(k0 + ar) * N + n0 + ac];
    __bf16* la = &As[ar * 64 + ac];
    __bf16* lb = &Bs[ar * 64 + ac];
    async_ld_b128(ga, la);
    async_ld_b128_off16(ga, la);
    async_ld_b128(gw, lb);
    async_ld_b128_off16(gw, lb);
    if (k0 + 64 < K) {
      __builtin_prefetch(&A[(size_t)(m0 + ar) * K + k0 + 64 + ac], 0, 1);
      __builtin_prefetch(&W[(size_t)(k0 + 64 + ar) * N + n0 + ac], 0, 1);
    }
    wait_async0();
    __syncthreads();
    v16bf a0  = frag_a(&As[mrow * 64], 64);
    v16bf a1  = frag_a(&As[mrow * 64 + 32], 64);
    v16bf b00 = frag_b(&Bs[ncol], 64);
    v16bf b01 = frag_b(&Bs[ncol + 16], 64);
    v16bf b10 = frag_b(&Bs[32 * 64 + ncol], 64);
    v16bf b11 = frag_b(&Bs[32 * 64 + ncol + 16], 64);
    acc0 = WMMA_BF16(a0, b00, acc0);
    acc1 = WMMA_BF16(a0, b01, acc1);
    acc0 = WMMA_BF16(a1, b10, acc0);
    acc1 = WMMA_BF16(a1, b11, acc1);
    __syncthreads();
  }
  const int cn = lane & 15, cm = (lane >> 4) * 8;
#pragma unroll
  for (int r = 0; r < 8; ++r) {
    const int row = m0 + mrow + cm + r;
    const int c0 = n0 + ncol + cn, c1 = c0 + 16;
    float v0 = acc0[r], v1 = acc1[r];
    if (bias) { v0 += bias[c0]; v1 += bias[c1]; }
    if (act == 1) { v0 = geluf(v0); v1 = geluf(v1); }
    const size_t o0 = (size_t)row * N + c0, o1 = (size_t)row * N + c1;
    if (Cf) { Cf[o0] = v0; Cf[o1] = v1; }
    if (Cb) { Cb[o0] = f2bf(v0); Cb[o1] = f2bf(v1); }
  }
}

// ---------------- attention scores: dots = (Q Kh^T)*scale + bias ----------------
__launch_bounds__(256)
__global__ void attn_qk_kernel(const __bf16* __restrict__ Q, const __bf16* __restrict__ Km,
                               const float* __restrict__ ts, float* __restrict__ dots) {
  __shared__ __align__(16) __bf16 Qs[64 * 96];
  __shared__ __align__(16) __bf16 Ks[96 * 64];
  const int tid = threadIdx.x, lane = tid & 31, w = tid >> 5;
  const int n0 = blockIdx.x * 64, h = blockIdx.y, b = blockIdx.z;
  for (int c = tid; c < 768; c += 256) {
    const int row = c / 12, c8 = (c % 12) * 8;
    async_ld_b128(&Q[((size_t)b * Lv + row) * Dv + h * HDv + c8], &Qs[row * 96 + c8]);
  }
  for (int c = tid; c < 768; c += 256) {
    const int key = c / 12, c8 = (c % 12) * 8;
    uint4 t = *(const uint4*)&Km[((size_t)b * Nv + n0 + key) * Dv + h * HDv + c8];
    const __bf16* te = reinterpret_cast<const __bf16*>(&t);
#pragma unroll
    for (int e = 0; e < 8; ++e) Ks[(c8 + e) * 64 + key] = te[e];
  }
  wait_async0();
  __syncthreads();
  const int mrow = (w >> 1) * 16, ncol = (w & 1) * 32;
  v8f acc0 = zero8(), acc1 = zero8();
#pragma unroll
  for (int kk = 0; kk < 96; kk += 32) {
    v16bf a  = frag_a(&Qs[mrow * 96 + kk], 96);
    v16bf b0 = frag_b(&Ks[kk * 64 + ncol], 64);
    v16bf b1 = frag_b(&Ks[kk * 64 + ncol + 16], 64);
    acc0 = WMMA_BF16(a, b0, acc0);
    acc1 = WMMA_BF16(a, b1, acc1);
  }
  const int cn = lane & 15, cm = (lane >> 4) * 8;
#pragma unroll
  for (int r = 0; r < 8; ++r) {
    const int l = mrow + cm + r;
    const size_t ro = (((size_t)b * Hv + h) * Lv + l) * Nv;
    const int j0c = n0 + ncol + cn;
    dots[ro + j0c]      = acc0[r] * kScale + ts[b * Nv + j0c];
    dots[ro + j0c + 16] = acc1[r] * kScale + ts[b * Nv + j0c + 16];
  }
}

// ---------------- row softmax over N=576 (one wave per row) -> bf16 P ----------------
__launch_bounds__(256)
__global__ void softmax_kernel(const float* __restrict__ dots, __bf16* __restrict__ P) {
  const int row = blockIdx.x * 8 + (threadIdx.x >> 5);
  const int lane = threadIdx.x & 31;
  const float* src = dots + (size_t)row * Nv;
  float m = -3e38f;
  for (int j = lane; j < Nv; j += 32) m = fmaxf(m, src[j]);
#pragma unroll
  for (int o = 16; o > 0; o >>= 1) m = fmaxf(m, __shfl_xor(m, o, 32));
  float s = 0.f;
  for (int j = lane; j < Nv; j += 32) s += expf(src[j] - m);
#pragma unroll
  for (int o = 16; o > 0; o >>= 1) s += __shfl_xor(s, o, 32);
  const float inv = 1.0f / s;
  __bf16* dst = P + (size_t)row * Nv;
  for (int j = lane; j < Nv; j += 32) dst[j] = f2bf(expf(src[j] - m) * inv);
}

// ---------------- P @ V per (b,h): out tile 64x96 -> [B,L,D] bf16 (heads re-interleaved) ----------------
__launch_bounds__(256)
__global__ void attn_av_kernel(const __bf16* __restrict__ P, const __bf16* __restrict__ Vm,
                               __bf16* __restrict__ Ob) {
  __shared__ __align__(16) __bf16 Ps[64 * 32];
  __shared__ __align__(16) __bf16 Vs[32 * 96];
  const int tid = threadIdx.x, lane = tid & 31, w = tid >> 5;
  const int h = blockIdx.x, b = blockIdx.y;
  const int mrow = (w & 3) * 16, c0 = (w >> 2) * 48;
  const int ar = tid >> 2, ac = (tid & 3) * 8;
  v8f acc[3] = {zero8(), zero8(), zero8()};
  for (int n0 = 0; n0 < Nv; n0 += 32) {
    async_ld_b128(&P[(((size_t)b * Hv + h) * Lv + ar) * Nv + n0 + ac], &Ps[ar * 32 + ac]);
    for (int c = tid; c < 384; c += 256) {
      const int r = c / 12, c8 = (c % 12) * 8;
      async_ld_b128(&Vm[((size_t)b * Nv + n0 + r) * Dv + h * HDv + c8], &Vs[r * 96 + c8]);
    }
    wait_async0();
    __syncthreads();
    v16bf a = frag_a(&Ps[mrow * 32], 32);
#pragma unroll
    for (int j = 0; j < 3; ++j) {
      v16bf bb = frag_b(&Vs[c0 + j * 16], 96);
      acc[j] = WMMA_BF16(a, bb, acc[j]);
    }
    __syncthreads();
  }
  const int cn = lane & 15, cm = (lane >> 4) * 8;
#pragma unroll
  for (int j = 0; j < 3; ++j)
#pragma unroll
    for (int r = 0; r < 8; ++r) {
      const int l = mrow + cm + r;
      const int col = h * HDv + c0 + j * 16 + cn;
      Ob[((size_t)b * Lv + l) * Dv + col] = f2bf(acc[j][r]);
    }
}

// ---------------- elementwise add (c may alias a) ----------------
__global__ void add_kernel(const float* __restrict__ a, const float* __restrict__ b,
                           float* __restrict__ c, int n) {
  const int i = blockIdx.x * 256 + threadIdx.x;
  if (i < n) c[i] = a[i] + b[i];
}

extern "C" void kernel_launch(void* const* d_in, const int* in_sizes, int n_in,
                              void* d_out, int out_size, void* d_ws, size_t ws_size,
                              hipStream_t stream) {
  (void)in_sizes; (void)n_in; (void)out_size; (void)ws_size;
  const float* x    = (const float*)d_in[0];
  const float* noise= (const float*)d_in[1];
  const float* n1g  = (const float*)d_in[2];
  const float* n1b  = (const float*)d_in[3];
  const float* sw   = (const float*)d_in[4];
  const float* sb   = (const float*)d_in[5];
  const float* wq   = (const float*)d_in[6];
  const float* wk   = (const float*)d_in[7];
  const float* wv   = (const float*)d_in[8];
  const float* pw   = (const float*)d_in[9];
  const float* pb   = (const float*)d_in[10];
  const float* n2g  = (const float*)d_in[11];
  const float* n2b  = (const float*)d_in[12];
  const float* f1w  = (const float*)d_in[13];
  const float* f1b  = (const float*)d_in[14];
  const float* f2w  = (const float*)d_in[15];
  const float* f2b  = (const float*)d_in[16];
  float* out = (float*)d_out;

  char* base = (char*)d_ws;
  size_t off = 0;
  auto carve = [&](size_t bytes) -> void* {
    void* p = base + off;
    off += (bytes + 255) & ~(size_t)255;
    return p;
  };
  const size_t BN = (size_t)Bv * Nv, BL = (size_t)Bv * Lv;
  float*    xn_f  = (float*)   carve(BN * Dv * 4);
  __bf16*   xn_b  = (__bf16*)  carve(BN * Dv * 2);
  float*    sqv   = (float*)   carve(BN * 4);
  float*    tsv   = (float*)   carve(BN * 4);
  float*    dm    = (float*)   carve(BN * Nv * 4);
  float*    dens  = (float*)   carve(BN * 4);
  unsigned* dmaxb = (unsigned*)carve(Bv * 4);
  float*    scor  = (float*)   carve(BN * 4);
  int*      idxd  = (int*)     carve(BL * 4);
  int*      idxc  = (int*)     carve(BN * 4);
  float*    qin_f = (float*)   carve(BL * Dv * 4);
  __bf16*   qin_b = (__bf16*)  carve(BL * Dv * 2);
  __bf16*   wq_b  = (__bf16*)  carve((size_t)Dv * Dv * 2);
  __bf16*   wk_b  = (__bf16*)  carve((size_t)Dv * Dv * 2);
  __bf16*   wv_b  = (__bf16*)  carve((size_t)Dv * Dv * 2);
  __bf16*   pw_b  = (__bf16*)  carve((size_t)Dv * Dv * 2);
  __bf16*   f1_b  = (__bf16*)  carve((size_t)Dv * 4 * Dv * 2);
  __bf16*   f2_b  = (__bf16*)  carve((size_t)Dv * 4 * Dv * 2);
  __bf16*   q_b   = (__bf16*)  carve(BL * Dv * 2);
  __bf16*   k_b   = (__bf16*)  carve(BN * Dv * 2);
  __bf16*   v_b   = (__bf16*)  carve(BN * Dv * 2);
  float*    dots  = (float*)   carve((size_t)Bv * Hv * Lv * Nv * 4);
  __bf16*   p_b   = (__bf16*)  carve((size_t)Bv * Hv * Lv * Nv * 2);
  __bf16*   ao_b  = (__bf16*)  carve(BL * Dv * 2);
  float*    prj_f = (float*)   carve(BL * Dv * 4);
  float*    fea_f = (float*)   carve(BL * Dv * 4);
  __bf16*   ln2_b = (__bf16*)  carve(BL * Dv * 2);
  __bf16*   h1_b  = (__bf16*)  carve(BL * 4 * Dv * 2);

  const int DD = Dv * Dv, D4D = Dv * 4 * Dv;

  // 1. LN1 + score + sum-of-squares
  ln1_kernel<<<(int)BN, 256, 0, stream>>>(x, n1g, n1b, sw, sb, xn_f, xn_b, sqv, tsv);
  // 2. weights -> bf16
  cvt_bf16_kernel<<<(DD + 255) / 256, 256, 0, stream>>>(wq, wq_b, DD);
  cvt_bf16_kernel<<<(DD + 255) / 256, 256, 0, stream>>>(wk, wk_b, DD);
  cvt_bf16_kernel<<<(DD + 255) / 256, 256, 0, stream>>>(wv, wv_b, DD);
  cvt_bf16_kernel<<<(DD + 255) / 256, 256, 0, stream>>>(pw, pw_b, DD);
  cvt_bf16_kernel<<<(D4D + 255) / 256, 256, 0, stream>>>(f1w, f1_b, D4D);
  cvt_bf16_kernel<<<(D4D + 255) / 256, 256, 0, stream>>>(f2w, f2_b, D4D);
  init_kernel<<<1, 64, 0, stream>>>(dmaxb);
  // 3. pairwise distances (WMMA)
  pairdist_kernel<<<dim3(Nv / 64, Nv / 64, Bv), 256, 0, stream>>>(xn_b, sqv, dm);
  // 4. DPC-KNN clustering
  density_kernel<<<(int)((BN + 255) / 256), 256, 0, stream>>>(dm, noise, dens, dmaxb);
  parent_kernel<<<(int)((BN + 255) / 256), 256, 0, stream>>>(dm, dens, dmaxb, scor);
  topl_kernel<<<Bv, 256, 0, stream>>>(scor, idxd);
  assign_kernel<<<(int)((BN + 255) / 256), 256, 0, stream>>>(dm, idxd, idxc);
  center_override_kernel<<<(int)((BL + 255) / 256), 256, 0, stream>>>(idxd, idxc);
  // 5. token merge -> q_input
  merge_kernel<<<dim3(Lv, Bv), 256, 0, stream>>>(xn_f, idxc, tsv, qin_f, qin_b);
  // 6. projections (WMMA, async-LDS staged)
  gemm_bf16_kernel<<<dim3(Dv / 64, (int)(BN / 64)), 256, 0, stream>>>(
      xn_b, wk_b, nullptr, nullptr, k_b, (int)BN, Dv, Dv, 0);
  gemm_bf16_kernel<<<dim3(Dv / 64, (int)(BN / 64)), 256, 0, stream>>>(
      xn_b, wv_b, nullptr, nullptr, v_b, (int)BN, Dv, Dv, 0);
  gemm_bf16_kernel<<<dim3(Dv / 64, (int)(BL / 64)), 256, 0, stream>>>(
      qin_b, wq_b, nullptr, nullptr, q_b, (int)BL, Dv, Dv, 0);
  // 7. attention (WMMA QK^T, softmax, WMMA PV)
  attn_qk_kernel<<<dim3(Nv / 64, Hv, Bv), 256, 0, stream>>>(q_b, k_b, tsv, dots);
  softmax_kernel<<<Bv * Hv * Lv / 8, 256, 0, stream>>>(dots, p_b);
  attn_av_kernel<<<dim3(Hv, Bv), 256, 0, stream>>>(p_b, v_b, ao_b);
  // 8. output projection + residual
  gemm_bf16_kernel<<<dim3(Dv / 64, (int)(BL / 64)), 256, 0, stream>>>(
      ao_b, pw_b, pb, prj_f, nullptr, (int)BL, Dv, Dv, 0);
  add_kernel<<<(int)(BL * Dv / 256), 256, 0, stream>>>(qin_f, prj_f, fea_f, (int)(BL * Dv));
  // 9. MLP
  ln2_kernel<<<(int)BL, 256, 0, stream>>>(fea_f, n2g, n2b, ln2_b);
  gemm_bf16_kernel<<<dim3(4 * Dv / 64, (int)(BL / 64)), 256, 0, stream>>>(
      ln2_b, f1_b, f1b, nullptr, h1_b, (int)BL, Dv, 4 * Dv, 1);
  gemm_bf16_kernel<<<dim3(Dv / 64, (int)(BL / 64)), 256, 0, stream>>>(
      h1_b, f2_b, f2b, out, nullptr, (int)BL, 4 * Dv, Dv, 0);
  // 10. final residual (in place on d_out)
  add_kernel<<<(int)(BL * Dv / 256), 256, 0, stream>>>(out, fea_f, out, (int)(BL * Dv));
}
